// MLPAttention_71829033058862
// MI455X (gfx1250) — matrix operand order repaired and checked by reading, not compiled, not run on hardware
//
#include <hip/hip_runtime.h>
#include <hip/hip_bf16.h>
#include <math.h>

// Problem constants (reference: B=32, Tq=Tk=1024, U=256)
#define B_  32
#define TQ_ 1024
#define TK_ 1024
#define U_  256

typedef __attribute__((ext_vector_type(16))) __bf16 v16bf;
typedef __attribute__((ext_vector_type(8)))  float  v8f;

// ---------------------------------------------------------------------------
// Kernel 1: sk[row] = sum_n tanh( (key @ W_q^T)[row,n] ) * v[n]
// rows = B*Tk = 32768, inner dims 256x256. One wave per 16-row tile,
// v_wmma_f32_16x16x32_bf16, fp32 accumulate, A block cached in VGPRs.
// ---------------------------------------------------------------------------
__global__ __launch_bounds__(256)
void MLPAttention_proj_score_kernel(const float* __restrict__ key,  // (B*Tk, U)
                                    const float* __restrict__ Wq,   // (U, U) row-major
                                    const float* __restrict__ vvec, // (U)
                                    float* __restrict__ sk)         // (B*Tk)
{
    const int lane = threadIdx.x & 31;
    const int wave = threadIdx.x >> 5;
    const int rowBlock = blockIdx.x * 8 + wave;   // 2048 tiles total (grid=256 x 8 waves)
    const int row0 = rowBlock * 16;

    const int m    = lane & 15;   // A row / C column index
    const int ksel = lane >> 4;   // lane half selects K sub-stripe

    // ---- preload entire 16x256 fp32 A block as bf16 into registers --------
    // A operand layout (16-bit, 16x32): lanes 0-15 hold K={k0..k0+7, k0+16..k0+23},
    // lanes 16-31 hold K={k0+8..k0+15, k0+24..k0+31}, for row M = lane&15.
    v16bf a[8];
    const float* arow = key + (size_t)(row0 + m) * U_;
    #pragma unroll
    for (int kt = 0; kt < 8; ++kt) {
        const float* p0 = arow + kt * 32 + ksel * 8;
        const float* p1 = p0 + 16;
        #pragma unroll
        for (int i = 0; i < 8; ++i) {
            a[kt][i]     = (__bf16)p0[i];
            a[kt][8 + i] = (__bf16)p1[i];
        }
    }

    float spart[8] = {0.f, 0.f, 0.f, 0.f, 0.f, 0.f, 0.f, 0.f};

    // ---- sweep the 16 N-tiles; 8-step K loop each -------------------------
    for (int nt = 0; nt < 16; ++nt) {
        const int n = nt * 16 + (lane & 15);
        // B operand (32x16): lane holds column N = lane&15; element i<8 maps to
        // K = ksel*8 + i, element 8+i to K = 16 + ksel*8 + i. B[k][n] = Wq[n][k]
        // (tk = key @ Wq^T), so we read two 8-float runs from row n of Wq per K-tile.
        const float* brow = Wq + (size_t)n * U_ + ksel * 8;
        v8f acc = {};
        #pragma unroll
        for (int kt = 0; kt < 8; ++kt) {
            const float* bp = brow + kt * 32;
            v16bf bm;
            #pragma unroll
            for (int i = 0; i < 8; ++i) { bm[i] = (__bf16)bp[i]; bm[8 + i] = (__bf16)bp[16 + i]; }
            acc = __builtin_amdgcn_wmma_f32_16x16x32_bf16(
                false, a[kt], false, bm, (short)0, acc, false, false);
        }
        // C/D layout: VGPR r -> M = r + 8*ksel, N = n (fixed per lane).
        const float vn = vvec[n];
        #pragma unroll
        for (int r = 0; r < 8; ++r) spart[r] += tanhf(acc[r]) * vn;
    }

    // ---- reduce over N: butterfly across the 16 lanes of each half --------
    #pragma unroll
    for (int r = 0; r < 8; ++r) {
        float s = spart[r];
        s += __shfl_xor(s, 1, 16);
        s += __shfl_xor(s, 2, 16);
        s += __shfl_xor(s, 4, 16);
        s += __shfl_xor(s, 8, 16);
        if ((lane & 15) == 0) sk[row0 + ksel * 8 + r] = s;
    }
}

// ---------------------------------------------------------------------------
// Kernel 2: masked softmax over k of sk[b,:]  (q-independent — see analysis).
// valid_length==0  -> all scores are NEG -> uniform 1/Tk (matches reference).
// valid_length> 0  -> masked lanes are exp(-1e6 - max) == 0.0f exactly.
// Also zeroes the o[] accumulator for kernel 3.
// ---------------------------------------------------------------------------
__global__ __launch_bounds__(1024)
void MLPAttention_softmax_kernel(const float* __restrict__ sk,
                                 const int* __restrict__ valid_length,
                                 float* __restrict__ attn,   // (B, Tk)
                                 float* __restrict__ o)      // (B, U) zero-init here
{
    __shared__ float red[TK_];
    const int b = blockIdx.x;
    const int k = threadIdx.x;
    const int vl = valid_length[b];
    const float s = sk[b * TK_ + k];
    const bool valid = (k < vl);

    red[k] = valid ? s : -3.4e38f;
    __syncthreads();
    for (int off = TK_ / 2; off > 0; off >>= 1) {
        if (k < off) red[k] = fmaxf(red[k], red[k + off]);
        __syncthreads();
    }
    const float mx = red[0];
    __syncthreads();

    const float e = (vl == 0) ? 1.0f : (valid ? __expf(s - mx) : 0.0f);
    red[k] = e;
    __syncthreads();
    for (int off = TK_ / 2; off > 0; off >>= 1) {
        if (k < off) red[k] += red[k + off];
        __syncthreads();
    }
    attn[b * TK_ + k] = e / red[0];

    if (k < U_) o[b * U_ + k] = 0.0f;
}

// ---------------------------------------------------------------------------
// Kernel 3: o[b,d] = sum_k attn[b,k] * value[b,k,d]   (rank-1, bandwidth-bound)
// grid (B, Tk/128); fp32 atomics across k-chunks, coalesced value reads.
// ---------------------------------------------------------------------------
__global__ __launch_bounds__(256)
void MLPAttention_av_kernel(const float* __restrict__ attn,
                            const float* __restrict__ value,
                            float* __restrict__ o)
{
    const int b  = blockIdx.x;
    const int k0 = blockIdx.y * 128;
    const int d  = threadIdx.x;
    const float* vb = value + ((size_t)b * TK_ + k0) * U_;
    const float* ab = attn + (size_t)b * TK_ + k0;
    float acc = 0.f;
    #pragma unroll 4
    for (int kk = 0; kk < 128; ++kk)
        acc += ab[kk] * vb[(size_t)kk * U_ + d];
    atomicAdd(&o[b * U_ + d], acc);
}

// ---------------------------------------------------------------------------
// Kernel 4: broadcast o[b,:] to out[b,q,:] for all q (pure store bandwidth).
// ---------------------------------------------------------------------------
__global__ __launch_bounds__(256)
void MLPAttention_bcast_kernel(const float* __restrict__ o,
                               float* __restrict__ out)
{
    const int b  = blockIdx.x;
    const int q0 = blockIdx.y * 32;
    const int d  = threadIdx.x;
    const float val = o[b * U_ + d];
    float* ob = out + ((size_t)b * TQ_ + q0) * U_ + d;
    #pragma unroll
    for (int r = 0; r < 32; ++r) ob[(size_t)r * U_] = val;
}

extern "C" void kernel_launch(void* const* d_in, const int* in_sizes, int n_in,
                              void* d_out, int out_size, void* d_ws, size_t ws_size,
                              hipStream_t stream) {
    // inputs: 0=query (algebraically unused), 1=key, 2=value, 3=valid_length,
    //         4=W_k (unused), 5=W_q, 6=v
    const float* key   = (const float*)d_in[1];
    const float* value = (const float*)d_in[2];
    const int*   vlen  = (const int*)d_in[3];
    const float* Wq    = (const float*)d_in[5];
    const float* vvec  = (const float*)d_in[6];
    float* out = (float*)d_out;

    // workspace: sk (B*Tk f32) | attn (B*Tk f32) | o (B*U f32)
    float* sk   = (float*)d_ws;
    float* attn = sk + (size_t)B_ * TK_;
    float* o    = attn + (size_t)B_ * TK_;

    // 2048 row-tiles of 16, 8 waves per 256-thread block -> 256 blocks
    MLPAttention_proj_score_kernel<<<256, 256, 0, stream>>>(key, Wq, vvec, sk);
    MLPAttention_softmax_kernel<<<B_, TK_, 0, stream>>>(sk, vlen, attn, o);
    MLPAttention_av_kernel<<<dim3(B_, TK_ / 128), 256, 0, stream>>>(attn, value, o);
    MLPAttention_bcast_kernel<<<dim3(B_, TQ_ / 32), 256, 0, stream>>>(o, out);
}
